// CausalSelfAttention_85031762526676
// MI455X (gfx1250) — compile-verified, matrix operand order checked by reading, not run on hardware
//
#include <hip/hip_runtime.h>

// ---------------------------------------------------------------------------
// Causal self-attention for MI455X (gfx1250, wave32, WMMA + TDM).
// Pipeline:  cast(x, Wqkv^T, Wproj^T -> f16)
//            -> WMMA GEMM (TDM double-buffered staging, qkv f16 out)
//            -> flash-attention (TDM K staging, WMMA QK^T / PV, online softmax)
//            -> WMMA GEMM (proj, f32 out + bias)
// ---------------------------------------------------------------------------

typedef _Float16 h8  __attribute__((ext_vector_type(8)));
typedef _Float16 h16 __attribute__((ext_vector_type(16)));
typedef float    f8  __attribute__((ext_vector_type(8)));
typedef unsigned int u32x4 __attribute__((ext_vector_type(4)));
typedef int          i32x4 __attribute__((ext_vector_type(4)));
typedef int          i32x8 __attribute__((ext_vector_type(8)));

static constexpr int Bsz  = 4;
static constexpr int Tseq = 2048;
static constexpr int Dmod = 1024;
static constexpr int Hn   = 16;
static constexpr int Hd   = 64;
static constexpr int Mrows = Bsz * Tseq;   // 8192
static constexpr int Nqkv  = 3 * Dmod;     // 3072

__device__ __forceinline__ f8 wmma16(h16 a, h16 b, f8 c) {
  // v_wmma_f32_16x16x32_f16  D = A*B + C
  return __builtin_amdgcn_wmma_f32_16x16x32_f16(false, a, false, b, (short)0, c,
                                                false, false);
}

// Load one 16x32 f16 WMMA A/B fragment from row-major storage.
// ISA layout (CDNA5 ISA): lane&15 -> row, lane>>4 selects K chunks
// {0..7,16..23} vs {8..15,24..31}. strideH must be a multiple of 8 halves.
__device__ __forceinline__ h16 load_frag(const _Float16* base, int strideH, int lane) {
  const int r = lane & 15;
  const int c = (lane >> 4) << 3;   // 0 or 8
  const _Float16* p = base + (size_t)r * strideH + c;
  h8 lo = *(const h8*)(p);
  h8 hi = *(const h8*)(p + 16);
  h16 f;
#pragma unroll
  for (int i = 0; i < 8; ++i) { f[i] = lo[i]; f[i + 8] = hi[i]; }
  return f;
}

// LDS byte offset of a __shared__ object: low 32 bits of the flat address
// (ISA 10.2, LDS aperture).
__device__ __forceinline__ unsigned lds_off(const void* p) {
  return (unsigned)(size_t)p;
}

// ---------------------------------------------------------------------------
// Tensor Data Mover: 2D f16 tile load Global -> LDS with optional row padding.
//   tile_d0   : contiguous halves per row
//   tile_d1   : rows
//   stride_d0 : row stride in memory (halves)
//   pad_interval/pad_amount per D# group1 encoding (CDNA5 ISA)
// tensor_dim set equal to tile_dim (tiles are always fully in-bounds here).
// ---------------------------------------------------------------------------
__device__ __forceinline__ void tdm_load_2d_f16(unsigned lds_byte_addr,
                                                const _Float16* gptr,
                                                unsigned tile_d0, unsigned tile_d1,
                                                unsigned stride_d0,
                                                unsigned pad_interval,
                                                unsigned pad_amount) {
  const unsigned long long ga = (unsigned long long)(size_t)gptr;
  u32x4 g0;
  g0[0] = 1u;                                   // count=1, user descriptor
  g0[1] = lds_byte_addr;                        // lds_addr
  g0[2] = (unsigned)ga;                         // global_addr[31:0]
  g0[3] = (unsigned)((ga >> 32) & 0x01FFFFFFull) | (2u << 30); // [56:32] | type=2
  i32x8 g1;
  g1[0] = (int)((1u << 16) |                    // data_size = 2 bytes
                (1u << 20) |                    // pad_enable
                (pad_interval << 22) | (pad_amount << 25));
  g1[1] = (int)((tile_d0 & 0xFFFFu) << 16);     // tensor_dim0[15:0]
  g1[2] = (int)(((tile_d0 >> 16) & 0xFFFFu) |   // tensor_dim0[31:16]
                ((tile_d1 & 0xFFFFu) << 16));   // tensor_dim1[15:0]
  g1[3] = (int)(((tile_d1 >> 16) & 0xFFFFu) |   // tensor_dim1[31:16]
                ((tile_d0 & 0xFFFFu) << 16));   // tile_dim0
  g1[4] = (int)(tile_d1 & 0xFFFFu);             // tile_dim1 (tile_dim2 = 0)
  g1[5] = (int)stride_d0;                       // tensor_dim0_stride[31:0]
  g1[6] = 0;                                    // stride0[47:32], dim1_stride lo
  g1[7] = 0;
  i32x4 z4 = {0, 0, 0, 0};
  i32x8 z8 = {0, 0, 0, 0, 0, 0, 0, 0};
  __builtin_amdgcn_tensor_load_to_lds(g0, g1, z4, z4, z8, 0);
}

// ---------------------------------------------------------------------------
// Cast helpers
// ---------------------------------------------------------------------------
__global__ void cast_f32_to_f16(const float* __restrict__ in,
                                _Float16* __restrict__ out, long n) {
  long i = (long)blockIdx.x * blockDim.x + threadIdx.x;
  if (i < n) out[i] = (_Float16)in[i];
}

// in: [K][N] f32 row-major  ->  out: [N][K] f16 row-major (transposed)
__global__ void cast_transpose_f16(const float* __restrict__ in,
                                   _Float16* __restrict__ out, int K, int N) {
  long i = (long)blockIdx.x * blockDim.x + threadIdx.x;
  if (i < (long)K * N) {
    int k = (int)(i / N), n = (int)(i % N);
    out[(long)n * K + k] = (_Float16)in[i];
  }
}

// ---------------------------------------------------------------------------
// WMMA GEMM: C[M,N] = A[M,K] * Bt[N,K]^T + bias[N]
// Block tile 128x128, 8 waves, wave tile 32x64 (2x4 wmma accumulators).
// TDM double-buffered LDS staging: wave 0 issues the DMA for tile k+1 while
// all waves compute tile k.
// ---------------------------------------------------------------------------
template <bool OUT_F32>
__global__ __launch_bounds__(256) void gemm_wmma(const _Float16* __restrict__ A,
                                                 const _Float16* __restrict__ Bt,
                                                 const float* __restrict__ bias,
                                                 void* __restrict__ Cout,
                                                 int M, int N, int K) {
  constexpr int LDT = 40;                     // padded LDS row stride (halves)
  constexpr unsigned BUFB = 128u * LDT * 2u;  // bytes per buffer (10240)
  __shared__ _Float16 As[2][128][LDT];
  __shared__ _Float16 Bs[2][128][LDT];

  const int tid  = threadIdx.x;
  const int wave = tid >> 5, lane = tid & 31;
  const int wm = wave >> 1, wn = wave & 1;
  const int m0 = blockIdx.y * 128, n0 = blockIdx.x * 128;

  const unsigned ldsA0 = lds_off(&As[0][0][0]);
  const unsigned ldsB0 = lds_off(&Bs[0][0][0]);

  f8 acc[2][4];
#pragma unroll
  for (int i = 0; i < 2; ++i)
#pragma unroll
    for (int j = 0; j < 4; ++j)
#pragma unroll
      for (int e = 0; e < 8; ++e) acc[i][j][e] = 0.0f;

  if (wave == 0) {
    // prime buffer 0:  A rows 32 halves (8x8B -> pad_interval=3), pad 4 DW
    tdm_load_2d_f16(ldsA0, A + (size_t)m0 * K, 32, 128, (unsigned)K, 3, 3);
    tdm_load_2d_f16(ldsB0, Bt + (size_t)n0 * K, 32, 128, (unsigned)K, 3, 3);
  }

  for (int k0 = 0; k0 < K; k0 += 32) {
    const int buf = (k0 >> 5) & 1;
    if (wave == 0) __builtin_amdgcn_s_wait_tensorcnt(0);
    __syncthreads();                          // tile k0 visible; buf^1 free
    if (wave == 0 && (k0 + 32) < K) {
      const int nb = buf ^ 1;
      tdm_load_2d_f16(ldsA0 + nb * BUFB, A + (size_t)m0 * K + k0 + 32,
                      32, 128, (unsigned)K, 3, 3);
      tdm_load_2d_f16(ldsB0 + nb * BUFB, Bt + (size_t)n0 * K + k0 + 32,
                      32, 128, (unsigned)K, 3, 3);
    }

    h16 af[2], bf[4];
#pragma unroll
    for (int i = 0; i < 2; ++i)
      af[i] = load_frag(&As[buf][wm * 32 + i * 16][0], LDT, lane);
#pragma unroll
    for (int j = 0; j < 4; ++j)
      bf[j] = load_frag(&Bs[buf][wn * 64 + j * 16][0], LDT, lane);
#pragma unroll
    for (int i = 0; i < 2; ++i)
#pragma unroll
      for (int j = 0; j < 4; ++j) acc[i][j] = wmma16(af[i], bf[j], acc[i][j]);
  }

  const int r = lane & 15, hf = lane >> 4;
#pragma unroll
  for (int i = 0; i < 2; ++i)
#pragma unroll
    for (int j = 0; j < 4; ++j)
#pragma unroll
      for (int e = 0; e < 8; ++e) {
        const int mr = m0 + wm * 32 + i * 16 + hf * 8 + e;
        const int nc = n0 + wn * 64 + j * 16 + r;
        const float v = acc[i][j][e] + bias[nc];
        if constexpr (OUT_F32) ((float*)Cout)[(size_t)mr * N + nc] = v;
        else ((_Float16*)Cout)[(size_t)mr * N + nc] = (_Float16)v;
      }
}

// ---------------------------------------------------------------------------
// Flash attention, causal. One block per (b, h, 128 q-rows); 8 waves,
// each wave owns 16 q-rows. K tiles of 32 columns; K staged by TDM,
// V staged transposed manually.
// ---------------------------------------------------------------------------
__global__ __launch_bounds__(256) void flash_attn(const _Float16* __restrict__ qkv,
                                                  _Float16* __restrict__ yh) {
  constexpr int BR = 128, BC = 32;
  constexpr int LK = 72;   // K tile row stride (halves): 32 rows x 64 d
  constexpr int LV = 40;   // V^T row stride: 64 d-rows x 32 t
  constexpr int LP = 40;   // P tile row stride: 16 rows x 32 cols
  __shared__ _Float16 Ks[BC][LK];
  __shared__ _Float16 Vt[Hd][LV];
  __shared__ _Float16 Ps[8][16][LP];

  const int qt = blockIdx.x, h = blockIdx.y, b = blockIdx.z;
  const int tid = threadIdx.x, wave = tid >> 5, lane = tid & 31;
  const int r = lane & 15, hf = lane >> 4;

  const _Float16* qbase = qkv + (size_t)b * Tseq * Nqkv + h * Hd;
  const _Float16* kbase = qbase + Dmod;
  const _Float16* vbase = qbase + 2 * Dmod;
  const unsigned ldsK = lds_off(&Ks[0][0]);

  const int qrow0 = qt * BR + wave * 16;
  const h16 qa0 = load_frag(qbase + (size_t)qrow0 * Nqkv,      Nqkv, lane); // d 0..31
  const h16 qa1 = load_frag(qbase + (size_t)qrow0 * Nqkv + 32, Nqkv, lane); // d 32..63

  f8 o[4];
  f8 mrow, lrow;
#pragma unroll
  for (int e = 0; e < 8; ++e) { mrow[e] = -1e30f; lrow[e] = 0.0f; }
#pragma unroll
  for (int dg = 0; dg < 4; ++dg)
#pragma unroll
    for (int e = 0; e < 8; ++e) o[dg][e] = 0.0f;

  const int njt = (qt + 1) * (BR / BC);
  for (int j = 0; j < njt; ++j) {
    const int kt0 = j * BC;
    __syncthreads();                      // previous tile fully consumed
    if (wave == 0) {
      // K tile: rows 64 halves (16x8B -> pad_interval=4), pad 4 DW -> LK=72
      tdm_load_2d_f16(ldsK, kbase + (size_t)kt0 * Nqkv, 64, 32,
                      (unsigned)Nqkv, 4, 3);
    }
    {
      const int row = tid >> 3;           // 0..31 (t within tile)
      const int ch  = (tid & 7) << 3;     // 0..56 (d chunk)
      h8 vv = *(const h8*)(vbase + (size_t)(kt0 + row) * Nqkv + ch);
#pragma unroll
      for (int e = 0; e < 8; ++e) Vt[ch + e][row] = vv[e];   // transpose V
    }
    if (wave == 0) __builtin_amdgcn_s_wait_tensorcnt(0);
    __syncthreads();

    if (qrow0 + 15 >= kt0) {              // wave-uniform: any unmasked element?
      // ---- S = Q K^T (16 x 32 scores, two 16-wide n-groups) ----
      f8 s[2];
#pragma unroll
      for (int ng = 0; ng < 2; ++ng) {
        h16 kb0 = load_frag(&Ks[ng * 16][0],      LK, lane); // d 0..31
        h16 kb1 = load_frag(&Ks[ng * 16][0] + 32, LK, lane); // d 32..63
        f8 a;
#pragma unroll
        for (int e = 0; e < 8; ++e) a[e] = 0.0f;
        a = wmma16(qa0, kb0, a);
        a = wmma16(qa1, kb1, a);
        s[ng] = a;
      }
      // ---- scale, causal mask, row max ----
      f8 mnew = mrow;
#pragma unroll
      for (int ng = 0; ng < 2; ++ng)
#pragma unroll
        for (int e = 0; e < 8; ++e) {
          const int col = kt0 + ng * 16 + r;
          const int row = qrow0 + hf * 8 + e;
          const float v = (col <= row) ? s[ng][e] * 0.125f : -1e30f;
          s[ng][e] = v;
          mnew[e] = fmaxf(mnew[e], v);
        }
#pragma unroll
      for (int off = 8; off >= 1; off >>= 1)
#pragma unroll
        for (int e = 0; e < 8; ++e)
          mnew[e] = fmaxf(mnew[e], __shfl_xor(mnew[e], off, 16));

      // ---- online softmax update ----
      f8 alpha, rsum;
#pragma unroll
      for (int e = 0; e < 8; ++e) { alpha[e] = __expf(mrow[e] - mnew[e]); rsum[e] = 0.0f; }
      mrow = mnew;
#pragma unroll
      for (int ng = 0; ng < 2; ++ng)
#pragma unroll
        for (int e = 0; e < 8; ++e) {
          const float p = __expf(s[ng][e] - mnew[e]);
          rsum[e] += p;
          Ps[wave][hf * 8 + e][ng * 16 + r] = (_Float16)p;  // C-layout -> A-layout
        }
#pragma unroll
      for (int off = 8; off >= 1; off >>= 1)
#pragma unroll
        for (int e = 0; e < 8; ++e) rsum[e] += __shfl_xor(rsum[e], off, 16);
#pragma unroll
      for (int e = 0; e < 8; ++e) lrow[e] = lrow[e] * alpha[e] + rsum[e];
#pragma unroll
      for (int dg = 0; dg < 4; ++dg)
#pragma unroll
        for (int e = 0; e < 8; ++e) o[dg][e] *= alpha[e];

      // ---- O += P V ----
      h16 pa = load_frag(&Ps[wave][0][0], LP, lane);
#pragma unroll
      for (int dg = 0; dg < 4; ++dg) {
        h16 vb = load_frag(&Vt[dg * 16][0], LV, lane);
        o[dg] = wmma16(pa, vb, o[dg]);
      }
    }
  }

  // ---- normalize and write y (f16, [B*T, D] with head-interleaved cols) ----
#pragma unroll
  for (int e = 0; e < 8; ++e) {
    const float inv = 1.0f / lrow[e];
    const int row = qrow0 + hf * 8 + e;
#pragma unroll
    for (int dg = 0; dg < 4; ++dg)
      yh[((size_t)b * Tseq + row) * Dmod + h * Hd + dg * 16 + r] =
          (_Float16)(o[dg][e] * inv);
  }
}

// ---------------------------------------------------------------------------
// Launch
// ---------------------------------------------------------------------------
extern "C" void kernel_launch(void* const* d_in, const int* in_sizes, int n_in,
                              void* d_out, int out_size, void* d_ws, size_t ws_size,
                              hipStream_t stream) {
  (void)in_sizes; (void)n_in; (void)out_size; (void)ws_size;
  const float* x      = (const float*)d_in[0];
  const float* w_attn = (const float*)d_in[1];
  const float* b_attn = (const float*)d_in[2];
  const float* w_proj = (const float*)d_in[3];
  const float* b_proj = (const float*)d_in[4];

  char* p = (char*)d_ws;
  _Float16* xh   = (_Float16*)p; p += (size_t)Mrows * Dmod * 2;   // 16.8 MB
  _Float16* wAT  = (_Float16*)p; p += (size_t)Nqkv  * Dmod * 2;   //  6.3 MB
  _Float16* wPT  = (_Float16*)p; p += (size_t)Dmod  * Dmod * 2;   //  2.1 MB
  _Float16* qkvh = (_Float16*)p; p += (size_t)Mrows * Nqkv * 2;   // 50.3 MB
  _Float16* yh   = (_Float16*)p; p += (size_t)Mrows * Dmod * 2;   // 16.8 MB

  {
    const long n = (long)Mrows * Dmod;
    cast_f32_to_f16<<<dim3((unsigned)((n + 255) / 256)), dim3(256), 0, stream>>>(x, xh, n);
  }
  cast_transpose_f16<<<dim3((unsigned)(((long)Dmod * Nqkv + 255) / 256)), dim3(256), 0, stream>>>(
      w_attn, wAT, Dmod, Nqkv);
  cast_transpose_f16<<<dim3((unsigned)(((long)Dmod * Dmod + 255) / 256)), dim3(256), 0, stream>>>(
      w_proj, wPT, Dmod, Dmod);

  gemm_wmma<false><<<dim3(Nqkv / 128, Mrows / 128), dim3(256), 0, stream>>>(
      xh, wAT, b_attn, qkvh, Mrows, Nqkv, Dmod);

  flash_attn<<<dim3(Tseq / 128, Hn, Bsz), dim3(256), 0, stream>>>(qkvh, yh);

  gemm_wmma<true><<<dim3(Dmod / 128, Mrows / 128), dim3(256), 0, stream>>>(
      yh, wPT, b_proj, d_out, Mrows, Dmod, Dmod);
}
